// NeuralHawkes_28707561407214
// MI455X (gfx1250) — matrix operand order repaired and checked by reading, not compiled, not run on hardware
//
#include <hip/hip_runtime.h>
#include <hip/hip_bf16.h>
#include <stdint.h>

// ---------------- problem constants ----------------
#define N_EV   4096
#define NDIMS_ 128
#define HDIM   1024
#define EDIM   512
#define MC_    64
#define GROWS  6144          // 6*H gate rows
#define XCOLS  1536          // E+H
#define NBLK   16            // persistent workgroups (one per WGP)

// ---------------- workspace layout (bytes) ----------------
// ctrl page: [0] barrier counter (u32), +64 lam_event (f32), +128 partial[32] (f32)
#define CTRL_OFF   0
#define HBF_OFF    4096                                   // h_t bf16 [1024]
#define GX_OFF     8192                                   // GX fp32 [129][6144]
#define GX_BYTES   (129*GROWS*4)
#define WHH_OFF    (GX_OFF + GX_BYTES)                    // W_hh bf16 A-tiles [384 MT][32 KT][32 lane][16]
#define WHH_BYTES  (GROWS*HDIM*2)
#define WINT_OFF   (WHH_OFF + WHH_BYTES)                  // W_int bf16 B-tiles [8 NT][32 KT][32 lane][16]
#define WINT_BYTES (NDIMS_*HDIM*2)
#define HSA_OFF    (WINT_OFF + WINT_BYTES)                // hs bf16 A-tiles [5 MT][32 KT][32 lane][16]
#define HSA_BYTES  (5*32*1024)

// ---------------- vector types ----------------
typedef __attribute__((ext_vector_type(16))) __bf16       v16bf;
typedef __attribute__((ext_vector_type(8)))  float        v8f;
typedef __attribute__((ext_vector_type(8)))  unsigned int v8u;
typedef __attribute__((ext_vector_type(4)))  unsigned int v4u;
typedef __attribute__((ext_vector_type(8)))  int          v8i;
typedef __attribute__((ext_vector_type(4)))  int          v4i;

// ---------------- helpers ----------------
__device__ __forceinline__ unsigned short f2bf(float f) {
  return __builtin_bit_cast(unsigned short, (__bf16)f);
}
__device__ __forceinline__ float sigmoidf_(float x) { return 1.f / (1.f + __expf(-x)); }
__device__ __forceinline__ float softplusf_(float x) { return (x > 20.f) ? x : log1pf(__expf(x)); }

__device__ __forceinline__ v8f wmma_bf16(v16bf a, v16bf b, v8f c) {
  return __builtin_amdgcn_wmma_f32_16x16x32_bf16(false, a, false, b, (short)0, c, false, false);
}

// Store one element into the 16-bit A-matrix VGPR-lane layout (ISA 7.12.2),
// tiles [MT][KT][lane 0..31][16 bf16], 32 K-tiles per M-tile row.
__device__ __forceinline__ void storeA(unsigned short* base, int m, int k, float v) {
  int mt = m >> 4, kt = k >> 5, ml = m & 15, klo = k & 31;
  int lane = ml + (((klo >> 3) & 1) << 4);
  int idx  = ((klo >> 4) << 3) + (klo & 7);
  base[(((long)(mt * 32 + kt)) * 32 + lane) * 16 + idx] = f2bf(v);
}

// Load one lane's 16 bf16 A/B fragment (32 bytes) as v16bf.
__device__ __forceinline__ v16bf ldA16(const unsigned short* p) {
  const v4u* q = (const v4u*)p;
  v4u a = q[0], b = q[1];
  v8u r = { a[0], a[1], a[2], a[3], b[0], b[1], b[2], b[3] };
  return __builtin_bit_cast(v16bf, r);
}

// Grid barrier: cluster split-barrier fast path (NOP when not cluster-dispatched)
// + deterministic monotonic atomic counter in L2.
__device__ __forceinline__ void grid_sync(unsigned* bar, unsigned& phase) {
  __syncthreads();
  if (threadIdx.x == 0) {
    __threadfence();
    asm volatile("s_barrier_signal -3" ::: "memory");
    phase += NBLK;
    __hip_atomic_fetch_add(bar, 1u, __ATOMIC_ACQ_REL, __HIP_MEMORY_SCOPE_AGENT);
    while (__hip_atomic_load(bar, __ATOMIC_ACQUIRE, __HIP_MEMORY_SCOPE_AGENT) < phase)
      __builtin_amdgcn_s_sleep(2);
    asm volatile("s_barrier_wait -3" ::: "memory");
    __threadfence();
  }
  __syncthreads();
}

// ================= kernel 1: weight swizzle (full GPU, one-time) =================
__global__ void nh_swizzle(const float* __restrict__ Wg, const float* __restrict__ Wi,
                           unsigned short* __restrict__ whh, unsigned short* __restrict__ wint) {
  const long total1 = (long)GROWS * HDIM;
  const long total2 = (long)NDIMS_ * HDIM;
  long stride = (long)gridDim.x * blockDim.x;
  for (long e = (long)blockIdx.x * blockDim.x + threadIdx.x; e < total1 + total2; e += stride) {
    if (e < total1) {                       // W_hh -> A-tile layout
      int row = (int)(e >> 10), k = (int)(e & 1023);
      storeA(whh, row, k, Wg[(long)row * XCOLS + EDIM + k]);
    } else {                                // W_int -> B-tile layout: B[k][n] = W_int[n][k]
      long e2 = e - total1;
      int n = (int)(e2 >> 10), k = (int)(e2 & 1023);
      int nt = n >> 4, kt = k >> 5;
      wint[(((long)(nt * 32 + kt)) * 32 + (n & 15) + (((k >> 3) & 1) << 4)) * 16 + (((k & 31) >> 4) << 3) + (k & 7)] = f2bf(Wi[(long)n * HDIM + k]);
    }
  }
}

// ============ kernel 2: GX[d][row] = W_x @ emb[d] + b_gates via WMMA ============
__global__ void nh_gx(const float* __restrict__ emb, const float* __restrict__ Wg,
                      const float* __restrict__ bg, float* __restrict__ gx) {
  int gwave = (blockIdx.x * blockDim.x + threadIdx.x) >> 5;
  int lane = threadIdx.x & 31;
  if (gwave >= 9 * 384) return;
  int mt = gwave / 384, nt = gwave % 384;
  int sel = lane >> 4, ml = lane & 15;
  int arow = mt * 16 + ml;                  // emb row for this lane (both lane halves)
  v8f acc = {};
  for (int kt = 0; kt < 16; ++kt) {
    v16bf A;
    if (arow < NDIMS_ + 1) {
      const float* p0 = emb + (long)arow * EDIM + kt * 32 + sel * 8;
      #pragma unroll
      for (int j = 0; j < 8; ++j) { A[j] = (__bf16)p0[j]; A[8 + j] = (__bf16)p0[16 + j]; }
    } else {
      #pragma unroll
      for (int j = 0; j < 16; ++j) A[j] = (__bf16)0.f;
    }
    v16bf B;                                // B[k][n] = W_gates[nbase+n][kbase+k], lane%16 = n
    int kcol = kt * 32 + sel * 8;
    #pragma unroll
    for (int j = 0; j < 16; ++j) B[j] = (__bf16)Wg[(long)(nt * 16 + ml) * XCOLS + kcol + (j & 7) + ((j >> 3) << 4)];
    acc = wmma_bf16(A, B, acc);
  }
  int col = nt * 16 + ml;
  float bias = bg[col];
  #pragma unroll
  for (int r = 0; r < 8; ++r) {
    int row = mt * 16 + r + (sel << 3);     // D: vgpr r = M=r (lanes 0-15) / M=r+8 (16-31)
    if (row < NDIMS_ + 1) gx[(long)row * GROWS + col] = acc[r] + bias;
  }
}

// ============ kernel 3: persistent sequential scan (16 WGPs) ============
__global__ __launch_bounds__(1024, 1)
void nh_seq(const float* __restrict__ times, const int* __restrict__ dims,
            const float* __restrict__ u, const float* __restrict__ bint,
            const int* __restrict__ tendp, char* __restrict__ ws, float* __restrict__ out) {
  const int tid = threadIdx.x, w = blockIdx.x;
  const int wave = tid >> 5, lane = tid & 31;

  unsigned* bar        = (unsigned*)(ws + CTRL_OFF);
  float* lam_event     = (float*)(ws + CTRL_OFF + 64);
  float* partial       = (float*)(ws + CTRL_OFF + 128);
  unsigned short* hbf  = (unsigned short*)(ws + HBF_OFF);
  const float* gx      = (const float*)(ws + GX_OFF);
  const unsigned short* whh = (const unsigned short*)(ws + WHH_OFF);
  unsigned short* hsa  = (unsigned short*)(ws + HSA_OFF);

  __shared__ v4u   s_wint[2048];            // 32 KB: this WGP's W_int B-panel (n-tile w&7)
  __shared__ float s_c[64], s_cb[64], s_dl[64], s_o[64], s_ct[64];
  __shared__ float s_gv[384];               // W_hh @ h_t for this WGP's gate rows

  // ---- TDM: stage W_int B-panel (32 KB) into LDS once ----
  if (tid == 0) {
    unsigned lds = (unsigned)(uintptr_t)&s_wint[0];
    unsigned long long ga = (unsigned long long)(uintptr_t)(ws + WINT_OFF)
                          + (unsigned long long)(w & 7) * 32768ull;
    v4u g0 = { 1u, lds, (unsigned)(ga & 0xffffffffu),
               ((unsigned)((ga >> 32) & 0x1ffffffu)) | 0x80000000u };   // type=2 (image)
    // data_size=8B, tensor_dim0=tile_dim0=4096, tensor_dim1=tile_dim1=1, stride0=4096
    v8i g1 = { (int)(3u << 16), (int)(4096u << 16), (int)(1u << 16),
               (int)(4096u << 16), 1, 4096, 0, 0 };
    v4i zv4 = { 0, 0, 0, 0 };
    v8i zv8 = { 0, 0, 0, 0, 0, 0, 0, 0 };
    __builtin_amdgcn_tensor_load_to_lds(g0, g1, zv4, zv4, zv8, 0);
    __builtin_amdgcn_s_wait_tensorcnt(0);
  }
  __syncthreads();

  float Tend = (float)tendp[0];

  // ---- BOS cell update: x = embedding[NDIMS], h = 0, c = 0 ----
  if (tid < 64) {
    int unit = w * 64 + tid;
    const float* g = gx + (long)NDIMS_ * GROWS;
    float ig = sigmoidf_(g[0 * HDIM + unit]);
    float og = sigmoidf_(g[2 * HDIM + unit]);
    float z  = tanhf(g[3 * HDIM + unit]);
    s_c[tid]  = ig * z;                                  // f * 0 + i * z
    s_cb[tid] = tanhf(g[4 * HDIM + unit]);
    s_dl[tid] = softplusf_(g[5 * HDIM + unit]);
    s_o[tid]  = og;
  }
  __syncthreads();

  unsigned phase = 0;
  float prev_t = 0.f, ll = 0.f, integ = 0.f;

  for (int n = 0; n <= N_EV; ++n) {
    const bool tail = (n == N_EV);
    int   dn = tail ? 0 : dims[n];
    float tn = tail ? Tend : times[n];
    float dt = fmaxf(tn - prev_t, 0.f);
    prev_t = tn;

    // ---- P1: MC samples hs[m][k] + event-time decay (this WGP's 64 units) ----
    {
      int k = tid & 63, grp = tid >> 6;
      float dl = s_dl[k], cc = s_c[k], cb = s_cb[k], oo = s_o[k];
      const float* ur = u + (long)n * MC_;
      #pragma unroll
      for (int j = 0; j < 4; ++j) {
        int m = grp * 4 + j;
        float s  = ur[m] * dt;
        float cs = cb + (cc - cb) * __expf(-s * dl);
        storeA(hsa, m, w * 64 + k, oo * tanhf(cs));
      }
      if (!tail && grp == 0) {
        float ct = cb + (cc - cb) * __expf(-dt * dl);
        float ht = oo * tanhf(ct);
        s_ct[k] = ct;
        hbf[w * 64 + k] = f2bf(ht);
        storeA(hsa, 64, w * 64 + k, ht);                 // row 64 = event-time intensity input
      }
    }
    grid_sync(bar, phase);

    // ---- P2: waves 0..23 -> W_hh matvec (WMMA), waves 24..31 -> intensity GEMM ----
    if (wave < 24) {
      if (!tail) {
        int MT = (wave >> 2) * 64 + w * 4 + (wave & 3); // gate g=wave/4, units w*64.., sub=wave%4
        v8f acc = {};
        for (int kt = 0; kt < 32; ++kt) {
          v16bf A = ldA16(whh + (((long)(MT * 32 + kt)) * 32 + lane) * 16);
          const v4u* hp = (const v4u*)(hbf + kt * 32 + ((lane >> 4) << 3)); // B: h_t replicated over 16 cols
          v4u h0 = hp[0], h1 = hp[2];
          v8u br = { h0[0], h0[1], h0[2], h0[3], h1[0], h1[1], h1[2], h1[3] };
          acc = wmma_bf16(A, __builtin_bit_cast(v16bf, br), acc);
        }
        if (lane == 0 || lane == 16) {                   // col 0 holds the 16 valid outputs
          int off = wave * 16 + ((lane >> 4) << 3);
          #pragma unroll
          for (int r = 0; r < 8; ++r) s_gv[off + r] = acc[r];
        }
      }
    } else {
      int slot = wave - 24;
      int t = w + 16 * slot;                             // output tile id, nt == w&7
      if (slot < 3 && t < 40) {
        int mt = t >> 3;
        v8f acc = {};
        for (int kt = 0; kt < 32; ++kt) {
          v16bf A = ldA16(hsa + (((long)(mt * 32 + kt)) * 32 + lane) * 16);
          const v4u* bp = s_wint + (size_t)(kt * 32 + lane) * 2;
          v4u b0 = bp[0], b1 = bp[1];
          v8u br = { b0[0], b0[1], b0[2], b0[3], b1[0], b1[1], b1[2], b1[3] };
          acc = wmma_bf16(A, __builtin_bit_cast(v16bf, br), acc);
        }
        int col = (w & 7) * 16 + (lane & 15);
        float bv = bint[col];
        if (mt < 4) {                                    // MC rows: softplus + tile sum
          float ps = 0.f;
          #pragma unroll
          for (int r = 0; r < 8; ++r) ps += softplusf_(acc[r] + bv);
          for (int o2 = 16; o2 > 0; o2 >>= 1) ps += __shfl_down(ps, o2, 32);
          if (lane == 0) partial[t] = ps;
        } else if (!tail) {                              // row 64: event intensity lam[d_n]
          if ((dn >> 4) == (w & 7) && lane == (dn & 15))
            *lam_event = softplusf_(acc[0] + bv);
        }
      }
    }
    grid_sync(bar, phase);

    // ---- P3: deterministic accumulate (WGP0) + gate nonlinearity / state update ----
    if (w == 0 && tid == 0) {
      float s = 0.f;
      for (int t2 = 0; t2 < 32; ++t2) s += partial[t2];
      integ += dt * s * (1.f / 64.f);
      if (!tail) ll += __logf(fmaxf(*lam_event, 1e-8f));
    }
    if (!tail && tid < 64) {
      int unit = w * 64 + tid;
      const float* g = gx + (long)dn * GROWS;            // GX includes W_x@emb + b_gates
      float gi = g[0 * HDIM + unit] + s_gv[0 * 64 + tid];
      float gf = g[1 * HDIM + unit] + s_gv[1 * 64 + tid];
      float go = g[2 * HDIM + unit] + s_gv[2 * 64 + tid];
      float gzv= g[3 * HDIM + unit] + s_gv[3 * 64 + tid];
      float gc = g[4 * HDIM + unit] + s_gv[4 * 64 + tid];
      float gd = g[5 * HDIM + unit] + s_gv[5 * 64 + tid];
      float ig = sigmoidf_(gi), fg = sigmoidf_(gf), og = sigmoidf_(go);
      s_c[tid]  = fg * s_ct[tid] + ig * tanhf(gzv);
      s_cb[tid] = tanhf(gc);
      s_dl[tid] = softplusf_(gd);
      s_o[tid]  = og;
    }
    __syncthreads();
  }

  if (w == 0 && tid == 0) out[0] = ll - integ;
}

// ================= host entry =================
extern "C" void kernel_launch(void* const* d_in, const int* in_sizes, int n_in,
                              void* d_out, int out_size, void* d_ws, size_t ws_size,
                              hipStream_t stream) {
  const float* times = (const float*)d_in[0];
  const int*   dims  = (const int*)d_in[1];
  const float* emb   = (const float*)d_in[2];
  const float* Wg    = (const float*)d_in[3];
  const float* bg    = (const float*)d_in[4];
  const float* Wi    = (const float*)d_in[5];
  const float* bi    = (const float*)d_in[6];
  const float* u     = (const float*)d_in[7];
  const int*   tend  = (const int*)d_in[8];
  float* out = (float*)d_out;
  char*  ws  = (char*)d_ws;

  (void)hipMemsetAsync(ws, 0, 4096, stream);   // reset barrier counter / ctrl page every launch

  nh_swizzle<<<2048, 256, 0, stream>>>(Wg, Wi,
      (unsigned short*)(ws + WHH_OFF), (unsigned short*)(ws + WINT_OFF));
  nh_gx<<<432, 256, 0, stream>>>(emb, Wg, bg, (float*)(ws + GX_OFF));
  nh_seq<<<NBLK, 1024, 0, stream>>>(times, dims, u, bi, tend, ws, out);
}
